// GAE_78297253806498
// MI455X (gfx1250) — matrix operand order repaired and checked by reading, not compiled, not run on hardware
//
#include <hip/hip_runtime.h>

// ---------------------------------------------------------------------------
// Types for WMMA (gfx1250, wave32)
// ---------------------------------------------------------------------------
typedef __attribute__((ext_vector_type(16))) __bf16          v16bf;
typedef __attribute__((ext_vector_type(8)))  float           v8f;
typedef __attribute__((ext_vector_type(8)))  unsigned short  ushort8;
typedef __attribute__((ext_vector_type(4)))  unsigned short  ushort4v;

struct Frag { ushort8 lo; ushort8 hi; };   // 32 bytes = 16 bf16 = one WMMA operand

__device__ __forceinline__ unsigned short f2bf(float f) {
    // round-to-nearest-even fp32 -> bf16
    unsigned int u = __float_as_uint(f);
    u += 0x7FFFu + ((u >> 16) & 1u);
    return (unsigned short)(u >> 16);
}

// ---------------------------------------------------------------------------
// CDNA5 async global->LDS copy (ASYNCcnt path), GVS addressing:
//   LDS[vdst + IOFF] = MEM[saddr + voff + IOFF]   (16 bytes per lane)
// Inline asm per CDNA5_HIP.md (portable across both toolchains).
// ---------------------------------------------------------------------------
template <int IOFF>
__device__ __forceinline__ void async_b128(unsigned lds, int voff, unsigned long long sbase) {
    asm volatile("global_load_async_to_lds_b128 %0, %1, %2 offset:%3"
                 :: "v"(lds), "v"(voff), "s"(sbase), "n"(IOFF) : "memory");
}
__device__ __forceinline__ void wait_async0() {
    asm volatile("s_wait_asynccnt 0x0" ::: "memory");
}

// ---------------------------------------------------------------------------
// Elementwise conversion kernels (HBM-bandwidth noise vs. the GEMMs)
// ---------------------------------------------------------------------------

// out[i] = bf16(c0*a[i] + c1*b[i]); n4 = n/4
__global__ void k_scale2_bf16(const float* __restrict__ a, const float* __restrict__ b,
                              unsigned short* __restrict__ out, float c0, float c1, long n4) {
    long stride = (long)gridDim.x * blockDim.x;
    for (long i = (long)blockIdx.x * blockDim.x + threadIdx.x; i < n4; i += stride) {
        float4 va = ((const float4*)a)[i];
        float4 vb = ((const float4*)b)[i];
        ushort4v o;
        o.x = f2bf(c0 * va.x + c1 * vb.x);
        o.y = f2bf(c0 * va.y + c1 * vb.y);
        o.z = f2bf(c0 * va.z + c1 * vb.z);
        o.w = f2bf(c0 * va.w + c1 * vb.w);
        ((ushort4v*)out)[i] = o;
    }
}

// out (M x Kp, bf16, row-major) = bf16(c0*z + c1*t), zero-padded cols [Ks, Kp)
__global__ void k_feat_bf16(const float* __restrict__ z, const float* __restrict__ t,
                            unsigned short* __restrict__ out,
                            int M, int Ks, int Kp, float c0, float c1) {
    long total = (long)M * Kp;
    long stride = (long)gridDim.x * blockDim.x;
    for (long i = (long)blockIdx.x * blockDim.x + threadIdx.x; i < total; i += stride) {
        int r = (int)(i / Kp), c = (int)(i % Kp);
        unsigned short v = 0;
        if (c < Ks) {
            long s = (long)r * Ks + c;
            v = f2bf(c0 * z[s] + c1 * t[s]);
        }
        out[i] = v;
    }
}

// out (Np x Kp, bf16) = transpose(bf16(W (Ks x Ns))), zero-padded.
// out[c*Kp + r] = W[r*Ns + c]  -> GEMM "B" operand stored column-major.
__global__ void k_wpadT_bf16(const float* __restrict__ W, unsigned short* __restrict__ out,
                             int Ks, int Ns, int Kp, int Np) {
    long total = (long)Np * Kp;
    long stride = (long)gridDim.x * blockDim.x;
    for (long i = (long)blockIdx.x * blockDim.x + threadIdx.x; i < total; i += stride) {
        int c = (int)(i / Kp), r = (int)(i % Kp);
        out[i] = (r < Ks && c < Ns) ? f2bf(W[(long)r * Ns + c]) : (unsigned short)0;
    }
}

// ---------------------------------------------------------------------------
// WMMA bf16 GEMM:  C(MxN) = A(MxK, bf16 row-major) * Bt(NxK, bf16 col-major)
//
// Block tile 128(M) x 256(N), BK=32, LDS double-buffered (one barrier/step).
// 8 waves as 2(M) x 4(N); wave tile 64x64 = 4x4 v_wmma_f32_16x16x32_bf16.
// Tiles are filled with global_load_async_to_lds_b128 (ASYNCcnt DMA path):
// no VGPR staging, no ds_store traffic -> DS pipe serves only fragment loads.
// K and the B buffer's column count are pre-padded -> guard-free hot loop,
// EXEC all-ones (WMMA requirement).
// Epilogue: optional fp32 (+ReLU) store, optional bf16 *transposed* store
// (feeds the next GEMM's column-major B operand; 16B contiguous per lane).
// ---------------------------------------------------------------------------
#define BM  128
#define BN  256
#define BK  32
#define LDK 40   // BK + 8 halves pad -> 80B pitch, 16B-aligned accesses

__global__ __launch_bounds__(256) void k_gemm_bf16(
    const unsigned short* __restrict__ A,  int lda,   // row pitch of A (elems)
    const unsigned short* __restrict__ Bt, int ldb,   // row pitch of Bt (= K pitch)
    float* __restrict__ Cf, int ldc,                  // fp32 out (nullable)
    unsigned short* __restrict__ CbT, int ldcb,       // bf16 transposed out (nullable)
    int N, int K, int relu) {

    __shared__ __align__(16) unsigned short sA[2][BM * LDK];  // 2 x 10 KiB
    __shared__ __align__(16) unsigned short sB[2][BN * LDK];  // 2 x 20 KiB

    const int tid  = threadIdx.x;
    const int lane = tid & 31;
    const int wave = tid >> 5;
    const int wm   = wave >> 2;          // 0..1 -> M offset 64*wm
    const int wn   = wave & 3;           // 0..3 -> N offset 64*wn
    const int bm   = blockIdx.y * BM;
    const int bn   = blockIdx.x * BN;

    // async-copy assignment: A: 2 x 16B per thread, B: 4 x 16B per thread
    const int ar = tid >> 1;             // A tile row 0..127
    const int ac = (tid & 1) * 16;       // A half-col {0,16}
    const int bc = tid;                  // B tile col 0..255 (full 32-half K row)

    // per-lane byte offsets into A / Bt (constant across the K loop)
    const int voffA = (int)(((size_t)(bm + ar) * lda + (size_t)ac) * 2);
    const int voffB = (int)((size_t)(bn + bc) * ldb * 2);
    // LDS byte addresses (low 32 bits of generic LDS address = LDS offset)
    unsigned ldsA2[2], ldsB2[2];
    ldsA2[0] = (unsigned)(size_t)&sA[0][ar * LDK + ac];
    ldsA2[1] = (unsigned)(size_t)&sA[1][ar * LDK + ac];
    ldsB2[0] = (unsigned)(size_t)&sB[0][bc * LDK];
    ldsB2[1] = (unsigned)(size_t)&sB[1][bc * LDK];

    auto copyAsync = [&](int p, int k0) {
        unsigned long long ga = (unsigned long long)(size_t)(A + k0);
        unsigned long long gb = (unsigned long long)(size_t)(Bt + k0);
        async_b128<0 >(ldsA2[p], voffA, ga);
        async_b128<16>(ldsA2[p], voffA, ga);
        async_b128<0 >(ldsB2[p], voffB, gb);
        async_b128<16>(ldsB2[p], voffB, gb);
        async_b128<32>(ldsB2[p], voffB, gb);
        async_b128<48>(ldsB2[p], voffB, gb);
    };

    v8f acc[4][4];
#pragma unroll
    for (int i = 0; i < 4; ++i)
#pragma unroll
        for (int j = 0; j < 4; ++j)
            acc[i][j] = v8f{0.f,0.f,0.f,0.f,0.f,0.f,0.f,0.f};

    copyAsync(0, 0);
    wait_async0();
    __syncthreads();

    const int nk  = K / BK;
    const int r15 = lane & 15;
    const int hi  = lane >> 4;
    int p = 0;

    for (int kk = 0; kk < nk; ++kk) {
        if (kk + 1 < nk) copyAsync(p ^ 1, (kk + 1) * BK);  // DMA next tile

        // A fragments: lane r15 = row; half-wave holds K {0-7,16-23}/{8-15,24-31}
        Frag fa[4], fb[4];
#pragma unroll
        for (int i = 0; i < 4; ++i) {
            const int row = wm * 64 + i * 16 + r15;
            fa[i].lo = *(const ushort8*)&sA[p][row * LDK + hi * 8];
            fa[i].hi = *(const ushort8*)&sA[p][row * LDK + hi * 8 + 16];
        }
        // B fragments: lane r15 = col; half-wave holds K {0-7,16-23}/{8-15,24-31}
#pragma unroll
        for (int j = 0; j < 4; ++j) {
            const int col = wn * 64 + j * 16 + r15;
            fb[j].lo = *(const ushort8*)&sB[p][col * LDK + hi * 8];
            fb[j].hi = *(const ushort8*)&sB[p][col * LDK + hi * 8 + 16];
        }

#pragma unroll
        for (int i = 0; i < 4; ++i)
#pragma unroll
            for (int j = 0; j < 4; ++j)
                acc[i][j] = __builtin_amdgcn_wmma_f32_16x16x32_bf16(
                    false, __builtin_bit_cast(v16bf, fa[i]),
                    false, __builtin_bit_cast(v16bf, fb[j]),
                    (short)0, acc[i][j], false, false);

        wait_async0();                           // our DMA writes are in LDS
        __syncthreads();                         // single barrier per K-step
        p ^= 1;
    }

    // Epilogue. C layout: VGPR v -> row = v + 8*(lane>=16), col = lane&15.
    const int rb8 = hi * 8;
#pragma unroll
    for (int i = 0; i < 4; ++i) {
#pragma unroll
        for (int j = 0; j < 4; ++j) {
            const int col  = bn + wn * 64 + j * 16 + r15;
            const int row0 = bm + wm * 64 + i * 16 + rb8;
            if (Cf && col < N) {
#pragma unroll
                for (int v = 0; v < 8; ++v) {
                    float val = acc[i][j][v];
                    if (relu) val = fmaxf(val, 0.0f);
                    Cf[(size_t)(row0 + v) * ldc + col] = val;
                }
            }
            if (CbT) {   // transposed bf16: 8 consecutive halves = one 16B store
                ushort8 o;
#pragma unroll
                for (int v = 0; v < 8; ++v) {
                    float val = acc[i][j][v];
                    if (relu) val = fmaxf(val, 0.0f);
                    o[v] = (col < N) ? f2bf(val) : (unsigned short)0;
                }
                *(ushort8*)(CbT + (size_t)col * ldcb + row0) = o;
            }
        }
    }
}

// ---------------------------------------------------------------------------
// Host orchestration
// ---------------------------------------------------------------------------
extern "C" void kernel_launch(void* const* d_in, const int* in_sizes, int n_in,
                              void* d_out, int out_size, void* d_ws, size_t ws_size,
                              hipStream_t stream) {
    (void)in_sizes; (void)n_in; (void)out_size; (void)ws_size;

    const float* x    = (const float*)d_in[0];
    const float* adj  = (const float*)d_in[1];
    const float* adjx = (const float*)d_in[2];
    const float* tra1 = (const float*)d_in[3];
    const float* tra2 = (const float*)d_in[4];
    const float* tra3 = (const float*)d_in[5];
    const float* h    = (const float*)d_in[6];
    const float* W1   = (const float*)d_in[7];
    const float* W2   = (const float*)d_in[8];
    const float* W3   = (const float*)d_in[9];
    const float* W4   = (const float*)d_in[10];
    const float* W5   = (const float*)d_in[11];

    const int N = 8192;
    float* out = (float*)d_out;                 // tuple order: z5,z1,z2,z3,z4
    float* z5 = out;
    float* z1 = z5 + (size_t)N * 10;
    float* z2 = z1 + (size_t)N * 500;
    float* z3 = z2 + (size_t)N * 500;
    float* z4 = z3 + (size_t)N * 2000;

    // Workspace:
    //   adjb  : 8192x8192 bf16 = 128 MiB (fits 192 MB L2 -> big GEMMs hit L2)
    //   featb : 8192x2048 bf16 =  32 MiB (row-major A of support GEMMs)
    //   supbT : 2048x8192 bf16 =  32 MiB (col-major B of big GEMMs)
    //   wbT   : up to 2048x512 bf16 = 2 MiB (col-major B of support GEMMs)
    char* ws = (char*)d_ws;
    unsigned short* adjb  = (unsigned short*)(ws);
    unsigned short* featb = (unsigned short*)(ws + (size_t)134217728);
    unsigned short* supbT = (unsigned short*)(ws + (size_t)167772160);
    unsigned short* wbT   = (unsigned short*)(ws + (size_t)201326592);

    const dim3 blk(256);
    const long nadj4 = (long)N * N / 4;

    auto gemm = [&](const unsigned short* A, int lda, const unsigned short* Bt, int ldb,
                    float* Cf, int ldc, unsigned short* CbT, int ldcb,
                    int n, int k, int relu) {
        dim3 grid((unsigned)((n + BN - 1) / BN), (unsigned)(N / BM));
        k_gemm_bf16<<<grid, blk, 0, stream>>>(A, lda, Bt, ldb, Cf, ldc, CbT, ldcb, n, k, relu);
    };

    // ---- Layer 1: z1 = relu(adj @ (x @ W1)) ----   Kp=1024, Np=512
    k_scale2_bf16<<<4096, blk, 0, stream>>>(adj, adj, adjb, 1.0f, 0.0f, nadj4);
    k_wpadT_bf16 <<<2048, blk, 0, stream>>>(W1, wbT, 1000, 500, 1024, 512);
    k_feat_bf16  <<<8192, blk, 0, stream>>>(x, x, featb, N, 1000, 1024, 1.0f, 0.0f);
    gemm(featb, 1024, wbT, 1024, nullptr, 0, supbT, 8192, 500, 1024, 0);
    gemm(adjb, 8192, supbT, 8192, z1, 500, nullptr, 0, 500, 8192, 1);

    // Blend adjacency once for layers 2..5 (overwrites adjb)
    k_scale2_bf16<<<4096, blk, 0, stream>>>(adj, adjx, adjb, 0.5f, 0.5f, nadj4);

    // ---- Layer 2: z2 = relu(adj_b @ ((.5 z1 + .5 tra1) @ W2)) ----  Kp=512, Np=512
    k_wpadT_bf16<<<1024, blk, 0, stream>>>(W2, wbT, 500, 500, 512, 512);
    k_feat_bf16 <<<8192, blk, 0, stream>>>(z1, tra1, featb, N, 500, 512, 0.5f, 0.5f);
    gemm(featb, 512, wbT, 512, nullptr, 0, supbT, 8192, 500, 512, 0);
    gemm(adjb, 8192, supbT, 8192, z2, 500, nullptr, 0, 500, 8192, 1);

    // ---- Layer 3: z3 = relu(adj_b @ ((.5 z2 + .5 tra2) @ W3)) ----  Kp=512, Np=2048
    k_wpadT_bf16<<<4096, blk, 0, stream>>>(W3, wbT, 500, 2000, 512, 2048);
    k_feat_bf16 <<<8192, blk, 0, stream>>>(z2, tra2, featb, N, 500, 512, 0.5f, 0.5f);
    gemm(featb, 512, wbT, 512, nullptr, 0, supbT, 8192, 2000, 512, 0);
    gemm(adjb, 8192, supbT, 8192, z3, 2000, nullptr, 0, 2000, 8192, 1);

    // ---- Layer 4: z4 = relu(adj_b @ ((.5 z3 + .5 tra3) @ W4)) ----  Kp=2048, Np=256
    k_wpadT_bf16<<<2048, blk, 0, stream>>>(W4, wbT, 2000, 10, 2048, 256);
    k_feat_bf16 <<<8192, blk, 0, stream>>>(z3, tra3, featb, N, 2000, 2048, 0.5f, 0.5f);
    gemm(featb, 2048, wbT, 2048, nullptr, 0, supbT, 8192, 10, 2048, 0);
    gemm(adjb, 8192, supbT, 8192, z4, 10, nullptr, 0, 10, 8192, 1);

    // ---- Layer 5: z5 = adj_b @ ((.5 z4 + .5 h) @ W5), no ReLU ----  Kp=32, Np=256
    k_wpadT_bf16<<<64,   blk, 0, stream>>>(W5, wbT, 10, 10, 32, 256);
    k_feat_bf16 <<<1024, blk, 0, stream>>>(z4, h, featb, N, 10, 32, 0.5f, 0.5f);
    gemm(featb, 32, wbT, 32, nullptr, 0, supbT, 8192, 10, 32, 0);
    gemm(adjb, 8192, supbT, 8192, z5, 10, nullptr, 0, 10, 8192, 0);
}
